// EstLossSepEmb_41884521071204
// MI455X (gfx1250) — matrix operand order repaired and checked by reading, not compiled, not run on hardware
//
#include <hip/hip_runtime.h>
#include <hip/hip_bf16.h>

typedef __attribute__((ext_vector_type(16))) _Float16 v16h;
typedef __attribute__((ext_vector_type(8)))  float    v8f;

#define BB     8192
#define DIM    256
#define EPSF   1e-8f
#define SLOT   (BB * DIM)        // elements per normalized f16 matrix
#define NSLICE 4                 // N-range slices per gemm
#define NT_PER_SLICE ((BB / 16) / NSLICE)   // 128 N-tiles per slice

struct alignas(16) H8 { _Float16 h[8]; };
struct alignas(16) F4 { float f[4]; };

// ---------------------------------------------------------------------------
// ws layout: 4 normalized f16 matrices (16 MB):
//   slot0 = v (vis_pred), slot1 = gt_v (vis_feats_proj),
//   slot2 = gt_t (caption_emb), slot3 = narr_v
// then: float acc[2] (loss sums), unsigned counts[3], pad,
//       float pmax[3*NSLICE*BB], int pidx[3*NSLICE*BB]  (partial argmax)
// ---------------------------------------------------------------------------

__global__ void init_acc_kernel(float* acc, unsigned* counts) {
    if (threadIdx.x == 0) {
        acc[0] = 0.0f; acc[1] = 0.0f;
        counts[0] = 0u; counts[1] = 0u; counts[2] = 0u;
    }
}

// One wave per (matrix, row): clamped-normalize fp32 row -> f16 row in ws.
__global__ void normalize_kernel(const float* __restrict__ p0,
                                 const float* __restrict__ p1,
                                 const float* __restrict__ p2,
                                 const float* __restrict__ p3,
                                 _Float16* __restrict__ ws16) {
    const int wave = threadIdx.x >> 5;
    const int lane = threadIdx.x & 31;
    const int row  = blockIdx.x * 8 + wave;
    const int slot = blockIdx.y;

    const float* src = (slot == 0) ? p0 : (slot == 1) ? p1 : (slot == 2) ? p2 : p3;

    const float* rp = src + (size_t)row * DIM + lane * 8;
    F4 a = *(const F4*)(rp);
    F4 b = *(const F4*)(rp + 4);

    float s = a.f[0]*a.f[0] + a.f[1]*a.f[1] + a.f[2]*a.f[2] + a.f[3]*a.f[3]
            + b.f[0]*b.f[0] + b.f[1]*b.f[1] + b.f[2]*b.f[2] + b.f[3]*b.f[3];
#pragma unroll
    for (int off = 16; off > 0; off >>= 1) s += __shfl_xor(s, off);

    const float inv = 1.0f / fmaxf(sqrtf(s), EPSF);

    H8 o;
#pragma unroll
    for (int j = 0; j < 4; ++j) o.h[j]     = (_Float16)(a.f[j] * inv);
#pragma unroll
    for (int j = 0; j < 4; ++j) o.h[4 + j] = (_Float16)(b.f[j] * inv);
    *(H8*)(ws16 + (size_t)slot * SLOT + (size_t)row * DIM + lane * 8) = o;
}

// One wave per row: two row-wise cosine losses, atomically accumulated.
__global__ void loss_kernel(const float* __restrict__ vis_pred,
                            const float* __restrict__ vis_feats,
                            const float* __restrict__ text_pred,
                            const float* __restrict__ cap_emb,
                            float* acc) {
    const int wave = threadIdx.x >> 5;
    const int lane = threadIdx.x & 31;
    const int row  = blockIdx.x * 8 + wave;
    const size_t base = (size_t)row * DIM + lane * 8;

    float d1 = 0.f, na1 = 0.f, nb1 = 0.f;
    float d2 = 0.f, na2 = 0.f, nb2 = 0.f;
#pragma unroll
    for (int half = 0; half < 2; ++half) {
        F4 a = *(const F4*)(vis_pred  + base + half * 4);
        F4 b = *(const F4*)(vis_feats + base + half * 4);
        F4 c = *(const F4*)(text_pred + base + half * 4);
        F4 d = *(const F4*)(cap_emb   + base + half * 4);
#pragma unroll
        for (int j = 0; j < 4; ++j) {
            d1 += a.f[j] * b.f[j]; na1 += a.f[j] * a.f[j]; nb1 += b.f[j] * b.f[j];
            d2 += c.f[j] * d.f[j]; na2 += c.f[j] * c.f[j]; nb2 += d.f[j] * d.f[j];
        }
    }
#pragma unroll
    for (int off = 16; off > 0; off >>= 1) {
        d1  += __shfl_xor(d1,  off); na1 += __shfl_xor(na1, off);
        nb1 += __shfl_xor(nb1, off); d2  += __shfl_xor(d2,  off);
        na2 += __shfl_xor(na2, off); nb2 += __shfl_xor(nb2, off);
    }
    if (lane == 0) {
        float c1 = d1 / (fmaxf(sqrtf(na1), EPSF) * fmaxf(sqrtf(nb1), EPSF));
        float c2 = d2 / (fmaxf(sqrtf(na2), EPSF) * fmaxf(sqrtf(nb2), EPSF));
        atomicAdd(&acc[0], 1.0f - c1);
        atomicAdd(&acc[1], 1.0f - c2);
    }
}

// grid = (128 row-blocks, NSLICE, 3 gemms), block = 128 (4 waves).
// Each wave owns one 16-row A tile in registers; all 4 waves sweep the SAME
// N-tile sequence (slice of 128 tiles) so the 8 KB B tile is served to all
// waves by the WGP$. 8 chained v_wmma_f32_16x16x32_f16 per tile (K=256),
// fused running argmax; per-row winners via xor-butterfly go straight to
// workspace partials (no LDS, no barriers).
__global__ void __launch_bounds__(128)
gemm_argmax_kernel(const _Float16* __restrict__ ws16,
                   float* __restrict__ pmax, int* __restrict__ pidx) {
    const int wave  = threadIdx.x >> 5;
    const int lane  = threadIdx.x & 31;
    const int half  = lane >> 4;     // 0: lanes 0-15, 1: lanes 16-31
    const int lr    = lane & 15;
    const int slice = blockIdx.y;
    const int g     = blockIdx.z;    // 0:gt_v  1:v  2:narr_v  (B always gt_t)
    const int mbase = blockIdx.x * 64 + wave * 16;

    const int aslot = (g == 0) ? 1 : (g == 1) ? 0 : 3;
    const _Float16* A = ws16 + (size_t)aslot * SLOT;
    const _Float16* T = ws16 + (size_t)2 * SLOT;   // gt_t

    // A tile, contiguous K order matching B: lane<16: halfs[0..15]=K(kb*32+0..15),
    // lane>=16: halfs[0..15]=K(kb*32+16..31).
    const _Float16* arow = A + (size_t)(mbase + lr) * DIM;
    v16h a[8];
#pragma unroll
    for (int kb = 0; kb < 8; ++kb) {
        const H8* lo = (const H8*)(arow + kb * 32 + half * 16);
        const H8* hi = (const H8*)(arow + kb * 32 + half * 16 + 8);
#pragma unroll
        for (int j = 0; j < 8; ++j) { a[kb][j] = lo->h[j]; a[kb][8 + j] = hi->h[j]; }
    }

    float bestv[8];
    int   besti[8];
#pragma unroll
    for (int r = 0; r < 8; ++r) { bestv[r] = -1e30f; besti[r] = 0x7fffffff; }

    const int ntBeg = slice * NT_PER_SLICE;
    const int ntEnd = ntBeg + NT_PER_SLICE;
    for (int nt = ntBeg; nt < ntEnd; ++nt) {   // uniform -> EXEC all 1s
        const int nbase = nt * 16;
        // B 32x16 in this code's contiguous K order (same as A): lane<16 holds K 0-15 of column N=lr -> 16 contiguous
        // halfs of the gt_t row (nbase+lr); lane>=16 holds K 16-31.
        const _Float16* trow = T + (size_t)(nbase + lr) * DIM;

        v8f c = {};
#pragma unroll
        for (int kb = 0; kb < 8; ++kb) {
            v16h b = *(const v16h*)(trow + kb * 32 + half * 16);
            c = __builtin_amdgcn_wmma_f32_16x16x32_f16(
                    false, a[kb], false, b, (short)0, c, false, false);
        }
        const int n = nbase + lr;   // D: component r -> (m = half*8+r, n = lr)
#pragma unroll
        for (int r = 0; r < 8; ++r) {
            float cv = c[r];
            if (cv > bestv[r]) { bestv[r] = cv; besti[r] = n; }
        }
    }

    // xor-butterfly reduce across the 16-lane group holding each row
#pragma unroll
    for (int off = 8; off > 0; off >>= 1) {
#pragma unroll
        for (int r = 0; r < 8; ++r) {
            float om = __shfl_xor(bestv[r], off);
            int   oi = __shfl_xor(besti[r], off);
            if (om > bestv[r] || (om == bestv[r] && oi < besti[r])) {
                bestv[r] = om; besti[r] = oi;
            }
        }
    }
    if (lr == 0) {
#pragma unroll
        for (int r = 0; r < 8; ++r) {
            const int row = mbase + half * 8 + r;
            const size_t k = ((size_t)(g * NSLICE + slice)) * BB + row;
            pmax[k] = bestv[r];
            pidx[k] = besti[r];
        }
    }
}

// Reduce NSLICE partials per (gemm, row); count argmax == row.
__global__ void count_kernel(const float* __restrict__ pmax,
                             const int* __restrict__ pidx,
                             unsigned* counts) {
    const int t = blockIdx.x * 256 + threadIdx.x;
    if (t >= 3 * BB) return;
    const int g = t / BB, row = t - g * BB;
    float bm = -1e30f;
    int   bi = 0x7fffffff;
#pragma unroll
    for (int s = 0; s < NSLICE; ++s) {   // ascending s => ascending n: strict >
        const size_t k = ((size_t)(g * NSLICE + s)) * BB + row;
        const float m = pmax[k];
        const int   i = pidx[k];
        if (m > bm || (m == bm && i < bi)) { bm = m; bi = i; }
    }
    if (bi == row) atomicAdd(&counts[g], 1u);
}

__global__ void finalize_kernel(const float* acc, const unsigned* counts, float* out) {
    if (threadIdx.x == 0) {
        const float inv = 1.0f / (float)BB;
        const float vl = acc[0] * inv;
        const float tl = acc[1] * inv;
        const float scale = 100.0f * inv;
        const float a_gtv  = (float)counts[0] * scale;
        const float a_v    = (float)counts[1] * scale;
        const float a_narr = (float)counts[2] * scale;
        out[0] = vl; out[1] = tl; out[2] = vl + tl;
        out[3] = a_gtv;  out[4] = a_v;
        out[5] = a_gtv;  out[6] = a_v;
        out[7] = a_narr; out[8] = a_narr;
    }
}

extern "C" void kernel_launch(void* const* d_in, const int* in_sizes, int n_in,
                              void* d_out, int out_size, void* d_ws, size_t ws_size,
                              hipStream_t stream) {
    const float* vis_pred  = (const float*)d_in[0];
    const float* text_pred = (const float*)d_in[1];
    const float* vis_feats = (const float*)d_in[2];
    const float* cap_emb   = (const float*)d_in[3];
    const float* narr      = (const float*)d_in[4];
    float* out = (float*)d_out;

    _Float16* ws16   = (_Float16*)d_ws;
    float*    acc    = (float*)(ws16 + (size_t)4 * SLOT);
    unsigned* counts = (unsigned*)(acc + 2);
    float*    pmax   = acc + 16;                   // 3*NSLICE*BB floats
    int*      pidx   = (int*)(pmax + 3 * NSLICE * BB);

    init_acc_kernel<<<1, 32, 0, stream>>>(acc, counts);

    // slot0=v(vis_pred) slot1=gt_v(vis_feats) slot2=gt_t(cap_emb) slot3=narr
    normalize_kernel<<<dim3(BB / 8, 4), 256, 0, stream>>>(
        vis_pred, vis_feats, cap_emb, narr, ws16);

    loss_kernel<<<BB / 8, 256, 0, stream>>>(vis_pred, vis_feats, text_pred, cap_emb, acc);

    gemm_argmax_kernel<<<dim3(BB / 64, NSLICE, 3), 128, 0, stream>>>(ws16, pmax, pidx);

    count_kernel<<<(3 * BB + 255) / 256, 256, 0, stream>>>(pmax, pidx, counts);

    finalize_kernel<<<1, 32, 0, stream>>>(acc, counts, out);
}
